// SGLSTMLayer_54357106098564
// MI455X (gfx1250) — compile-verified, matrix operand order checked by reading, not run on hardware
//
#include <hip/hip_runtime.h>
#include <hip/hip_bf16.h>

// ---------------------------------------------------------------------------
// Stacked LSTM (B=32, S=2048, D=H=768, L=3) for MI455X (gfx1250, wave32, WMMA)
//
//  lstm_xgemm: xg[s,b,4H] = x[b,s,:] @ Wx^T + Wx_b + bias  (bf16 WMMA GEMM,
//      float4 loads + native __bf16 conversion + ds_store_b64 staging).
//  lstm_scan: persistent 24-WG kernel; Wh slice (768x128 bf16 = 192KB) is
//      LDS-resident for the whole layer; h ping-pongs through global bf16
//      buffers, restaged each step with GLOBAL_LOAD_ASYNC_TO_LDS_B128
//      (ASYNCcnt) + s_wait_asynccnt; one agent-scope barrier per timestep;
//      next-step xg slice prefetched with global_prefetch_b8.
// ---------------------------------------------------------------------------

#define BATCH   32
#define SEQ     2048
#define DIN     768
#define HID     768
#define GATES   3072            // 4*HID
#define LAYERS  3
#define NWG     24              // HID / WGCOLS
#define WGCOLS  32              // h columns owned per workgroup

typedef __attribute__((ext_vector_type(16))) __bf16          v16bf;
typedef __attribute__((ext_vector_type(2)))  __bf16          v2bf;
typedef __attribute__((ext_vector_type(8)))  float           v8f;
typedef __attribute__((ext_vector_type(8)))  unsigned short  ushort8;

union BFrag { v16bf v; ushort8 u[2]; };

__device__ __forceinline__ unsigned short f2bf(float f) {
  unsigned u = __float_as_uint(f);
  unsigned r = 0x7FFFu + ((u >> 16) & 1u);       // round-to-nearest-even
  return (unsigned short)((u + r) >> 16);
}

#if __has_builtin(__builtin_amdgcn_cvt_pk_bf16_f32)
__device__ __forceinline__ unsigned long long pack4bf(float4 v) {
  union { v2bf v; unsigned u; } a, b;
  a.v = __builtin_amdgcn_cvt_pk_bf16_f32(v.x, v.y);
  b.v = __builtin_amdgcn_cvt_pk_bf16_f32(v.z, v.w);
  return ((unsigned long long)b.u << 32) | (unsigned long long)a.u;
}
#else
// native __bf16 truncation: lets the backend select its packed/scalar
// v_cvt_*bf16_f32 instructions (worst case it expands inline, same cost)
__device__ __forceinline__ unsigned long long pack4bf(float4 v) {
  union { __bf16 h[4]; unsigned long long u; } p;
  p.h[0] = (__bf16)v.x;
  p.h[1] = (__bf16)v.y;
  p.h[2] = (__bf16)v.z;
  p.h[3] = (__bf16)v.w;
  return p.u;
}
#endif

__device__ __forceinline__ float sigf(float x)  { return 1.0f / (1.0f + __expf(-x)); }
__device__ __forceinline__ float tanhf_(float x){ return 2.0f * sigf(2.0f * x) - 1.0f; }

// ---------------------------------------------------------------------------
// zero h/c state + barrier counters once per call (deterministic)
// ---------------------------------------------------------------------------
__global__ void lstm_init_state(float* __restrict__ c,
                                unsigned short* __restrict__ hbf,
                                unsigned* __restrict__ bar) {
  int i = blockIdx.x * blockDim.x + threadIdx.x;
  const int n = BATCH * HID;
  if (i < n)      c[i]   = 0.0f;
  if (i < 2 * n)  hbf[i] = 0;
  if (i < 2)      bar[i] = 0u;
}

// ---------------------------------------------------------------------------
// Input projection GEMM: XG[s, b, 0:4H] = X[b*S+s, :] @ W^T + Wb + Bb
//   X : [BATCH*SEQ, DIN] fp32 (row = b*SEQ + s)
//   W : [GATES, DIN]     fp32 (layer slice)
//   XG: [SEQ, BATCH, GATES] fp32 (transposed for contiguous per-step reads)
// Block tile: 64 rows x 128 cols, 8 waves (2x4), each wave 32x32 (2x2 WMMA).
// ---------------------------------------------------------------------------
#define GM_AS 40
#define GM_BS 40

__global__ __launch_bounds__(256, 2)
void lstm_xgemm(const float* __restrict__ X,
                const float* __restrict__ W,
                const float* __restrict__ Wb,
                const float* __restrict__ Bb,
                float* __restrict__ XG) {
  __shared__ unsigned short As[64  * GM_AS];
  __shared__ unsigned short Bs[128 * GM_BS];

  const int tid   = threadIdx.x;
  const int lane  = tid & 31;
  const int l15   = lane & 15;
  const int lhi   = lane >> 4;
  const int w8    = tid >> 5;
  const int waveM = w8 >> 2;                    // 0..1 (rows)
  const int waveN = w8 & 3;                    // 0..3 (cols)
  const long rowBase = (long)blockIdx.y * 64;   // over BATCH*SEQ
  const int  colBase = blockIdx.x * 128;        // over GATES
  // 64-row tile never crosses a batch boundary (64 | SEQ)
  const long bb0 = rowBase / SEQ;
  const int  ss0 = (int)(rowBase - bb0 * SEQ);

  v8f acc[2][2];
  const v8f vz = {0.f,0.f,0.f,0.f,0.f,0.f,0.f,0.f};
  acc[0][0] = vz; acc[0][1] = vz; acc[1][0] = vz; acc[1][1] = vz;

  for (int kk = 0; kk < DIN; kk += 32) {
    // stage A (64x32): float4 load -> bf16x4 -> ds_store_b64
#pragma unroll
    for (int i = tid; i < 64 * 8; i += 256) {
      int r = i >> 3, c4 = (i & 7) * 4;
      float4 v = *(const float4*)&X[(rowBase + r) * DIN + kk + c4];
      *(unsigned long long*)&As[r * GM_AS + c4] = pack4bf(v);
    }
    // stage B (128x32): rows are output gate columns of W (row-major [4H][D])
#pragma unroll
    for (int i = tid; i < 128 * 8; i += 256) {
      int r = i >> 3, c4 = (i & 7) * 4;
      float4 v = *(const float4*)&W[(long)(colBase + r) * DIN + kk + c4];
      *(unsigned long long*)&Bs[r * GM_BS + c4] = pack4bf(v);
    }
    __syncthreads();

    BFrag a[2], b[2];
#pragma unroll
    for (int mt = 0; mt < 2; ++mt) {
      const unsigned short* p = As + (waveM * 32 + mt * 16 + l15) * GM_AS + lhi * 8;
      a[mt].u[0] = *(const ushort8*)(p);
      a[mt].u[1] = *(const ushort8*)(p + 16);
    }
#pragma unroll
    for (int nt = 0; nt < 2; ++nt) {
      const unsigned short* p = Bs + (waveN * 32 + nt * 16 + l15) * GM_BS + lhi * 8;
      b[nt].u[0] = *(const ushort8*)(p);
      b[nt].u[1] = *(const ushort8*)(p + 16);
    }
#pragma unroll
    for (int mt = 0; mt < 2; ++mt)
#pragma unroll
      for (int nt = 0; nt < 2; ++nt)
        acc[mt][nt] = __builtin_amdgcn_wmma_f32_16x16x32_bf16(
            false, a[mt].v, false, b[nt].v, (short)0, acc[mt][nt], false, false);
    __syncthreads();
  }

  // epilogue: add biases, store to XG transposed [SEQ, BATCH, GATES]
#pragma unroll
  for (int mt = 0; mt < 2; ++mt) {
#pragma unroll
    for (int nt = 0; nt < 2; ++nt) {
      int n = colBase + waveN * 32 + nt * 16 + l15;
      float bn = Wb[n] + Bb[n];
#pragma unroll
      for (int i = 0; i < 8; ++i) {
        int ml = waveM * 32 + mt * 16 + lhi * 8 + i;
        XG[((long)(ss0 + ml) * BATCH + bb0) * (long)GATES + n] = acc[mt][nt][i] + bn;
      }
    }
  }
}

// ---------------------------------------------------------------------------
// grid-wide generation barrier (agent scope)
// ---------------------------------------------------------------------------
__device__ __forceinline__ void gridBarrier(unsigned* bar, unsigned nwg) {
  __syncthreads();
  if (threadIdx.x == 0) {
    __threadfence();
    unsigned g = __hip_atomic_load(&bar[1], __ATOMIC_RELAXED, __HIP_MEMORY_SCOPE_AGENT);
    unsigned a = __hip_atomic_fetch_add(&bar[0], 1u, __ATOMIC_ACQ_REL, __HIP_MEMORY_SCOPE_AGENT);
    if (a == nwg - 1) {
      __hip_atomic_store(&bar[0], 0u, __ATOMIC_RELAXED, __HIP_MEMORY_SCOPE_AGENT);
      __hip_atomic_fetch_add(&bar[1], 1u, __ATOMIC_RELEASE, __HIP_MEMORY_SCOPE_AGENT);
    } else {
      while (__hip_atomic_load(&bar[1], __ATOMIC_ACQUIRE, __HIP_MEMORY_SCOPE_AGENT) == g)
        __builtin_amdgcn_s_sleep(4);
    }
    __threadfence();
  }
  __syncthreads();
}

// ---------------------------------------------------------------------------
// Persistent recurrent scan. 24 WGs x 256 threads (8 waves).
// Dynamic LDS layout (bytes):
//   wh : 128 x 776 ushort  (Wh slice, bf16, padded K stride)   = 198656
//   hb : 32  x 776 ushort  (current h, bf16)                   =  49664
//   gb : 32  x 132 float   (gate exchange buffer)              =  16896
//   cb : 32  x 33  float   (cell-state slice)                  =   4224
// total = 269440 bytes (<= 320KB WGP LDS)
// ---------------------------------------------------------------------------
#define WH_S 776
#define HB_S 776
#define GB_S 132
#define CB_S 33
#define SCAN_LDS 269440

// Stage h (32x768 bf16, 48KB) global -> LDS with async b128 copies (ASYNCcnt).
// 3072 16-byte chunks, 12 per thread. LDS dest offset = low 32 bits of the
// generic LDS pointer (flat->LDS mapping uses addr[31:0]).
__device__ __forceinline__ void asyncStageH(const unsigned short* __restrict__ src,
                                            unsigned short* hb, int tid) {
  unsigned hbBase = (unsigned)(unsigned long long)hb;
#pragma unroll
  for (int r = 0; r < 12; ++r) {
    int chunk = tid + r * 256;
    int m = chunk / 96;                       // 96 chunks of 8 ushorts per row
    int k = (chunk - m * 96) * 8;
    unsigned ldsOff = hbBase + (unsigned)(m * HB_S + k) * 2u;
    unsigned long long ga = (unsigned long long)(src + chunk * 8);
    asm volatile("global_load_async_to_lds_b128 %0, %1, off"
                 :: "v"(ldsOff), "v"(ga) : "memory");
  }
  asm volatile("s_wait_asynccnt 0x0" ::: "memory");
}

__global__ __launch_bounds__(256, 1)
void lstm_scan(const float* __restrict__ XG,      // [SEQ, BATCH, GATES]
               const float* __restrict__ Wh,      // [GATES, HID] layer slice
               const float* __restrict__ Whb,     // [GATES]
               float* __restrict__ Y,             // [BATCH, SEQ, HID]
               float* __restrict__ Cst,           // [BATCH, HID]
               unsigned short* __restrict__ Hbf,  // [2][BATCH*HID] bf16 ping-pong
               unsigned* __restrict__ bar) {
  extern __shared__ char smem[];
  unsigned short* wh = (unsigned short*)smem;         // 128 * WH_S
  unsigned short* hb = wh + 128 * WH_S;               // 32 * HB_S
  float*          gb = (float*)(hb + 32 * HB_S);      // 32 * GB_S
  float*          cb = gb + 32 * GB_S;                // 32 * CB_S

  const int tid  = threadIdx.x;
  const int lane = tid & 31;
  const int l15  = lane & 15;
  const int lhi  = lane >> 4;
  const int w8   = tid >> 5;
  const int wg   = blockIdx.x;                        // owns h cols [wg*32, wg*32+32)

  // ---- preload Wh slice (float4 load -> bf16x4 -> ds_store_b64) ------------
  for (int i = tid; i < 128 * (HID / 4); i += 256) {
    int gr = i / (HID / 4), c4 = (i - gr * (HID / 4)) * 4;
    int gate = gr >> 5, j = gr & 31;
    float4 v = *(const float4*)&Wh[((long)gate * HID + wg * WGCOLS + j) * (long)HID + c4];
    *(unsigned long long*)&wh[gr * WH_S + c4] = pack4bf(v);
  }
  // ---- preload c slice -----------------------------------------------------
#pragma unroll
  for (int r = 0; r < 4; ++r) {
    int idx = tid + r * 256;
    int m = idx >> 5, j = idx & 31;
    cb[m * CB_S + j] = Cst[m * HID + wg * WGCOLS + j];
  }
  // ---- per-tile constants, invariant across all timesteps ------------------
  int   GcR[2];
  float whbR[2];
#pragma unroll
  for (int rep = 0; rep < 2; ++rep) {
    int tile = w8 * 2 + rep;
    int nt   = tile >> 1;
    int gcol = nt * 16 + l15;
    int gate = gcol >> 5, j = gcol & 31;
    GcR[rep]  = gate * HID + wg * WGCOLS + j;        // global gate column
    whbR[rep] = Whb[GcR[rep]];                       // hoisted Wh bias
  }
  // ---- initial h stage from ping-pong buffer 0 (async -> LDS) --------------
  asyncStageH(Hbf, hb, tid);
  __syncthreads();

  for (int t = 0; t < SEQ; ++t) {
    const float* xgT = XG + ((long)t * BATCH) * (long)GATES;
    // ---- gates = h @ Wh_slice^T  (16 tiles, 2 per wave) --------------------
#pragma unroll
    for (int rep = 0; rep < 2; ++rep) {
      const int tile  = w8 * 2 + rep;
      const int mtile = tile & 1;                    // batch-half
      const int nt    = tile >> 1;                   // 0..7 over 128 gate cols
      const int mb    = mtile * 16 + lhi * 8;
      const unsigned short* aB = hb + (mtile * 16 + l15) * HB_S + lhi * 8;
      const unsigned short* bB = wh + (nt    * 16 + l15) * WH_S + lhi * 8;
      // issue xg_t loads early; latency hides under the 24 WMMAs below
      const float* xgrow = xgT + GcR[rep];
      float xv[8];
#pragma unroll
      for (int i = 0; i < 8; ++i)
        xv[i] = xgrow[(long)(mb + i) * GATES];

      v8f acc = {0.f,0.f,0.f,0.f,0.f,0.f,0.f,0.f};
#pragma unroll
      for (int kt = 0; kt < HID / 32; ++kt) {
        BFrag a, b;
        a.u[0] = *(const ushort8*)(aB + kt * 32);
        a.u[1] = *(const ushort8*)(aB + kt * 32 + 16);
        b.u[0] = *(const ushort8*)(bB + kt * 32);
        b.u[1] = *(const ushort8*)(bB + kt * 32 + 16);
        acc = __builtin_amdgcn_wmma_f32_16x16x32_bf16(
            false, a.v, false, b.v, (short)0, acc, false, false);
      }
      // gates = wmma + xg_t + Wh_b, spill to gate exchange buffer
      const int gcol = nt * 16 + l15;                // 0..127 within slice
      const float whb = whbR[rep];
#pragma unroll
      for (int i = 0; i < 8; ++i)
        gb[(mb + i) * GB_S + gcol] = acc[i] + whb + xv[i];
    }
    __syncthreads();

    // ---- LSTM nonlinearity on this WG's 32x32 slice ------------------------
    unsigned short* hnext = Hbf + ((t + 1) & 1) * (BATCH * HID);
#pragma unroll
    for (int r = 0; r < 4; ++r) {
      int idx = tid + r * 256;
      int m = idx >> 5, j = idx & 31;
      float gi = gb[m * GB_S +       j];
      float gf = gb[m * GB_S + 32  + j];
      float gg = gb[m * GB_S + 64  + j];
      float go = gb[m * GB_S + 96  + j];
      float c  = sigf(gf) * cb[m * CB_S + j] + sigf(gi) * tanhf_(gg);
      cb[m * CB_S + j] = c;
      float h  = sigf(go) * tanhf_(c);
      int col = wg * WGCOLS + j;
      Y[((long)m * SEQ + t) * (long)HID + col] = h;
      hnext[m * HID + col] = f2bf(h);
    }
    // prefetch next timestep's xg slice: one 128B line per (m, gate)
    if (t + 1 < SEQ && tid < 128) {
      int m = tid >> 2, gate = tid & 3;
      __builtin_prefetch(&XG[((long)(t + 1) * BATCH + m) * (long)GATES +
                             gate * HID + wg * WGCOLS], 0, 1);
    }

    // ---- all WGs published h(t+1): one barrier (ping-pong makes it safe) ---
    gridBarrier(bar, NWG);

    // ---- restage full h(t+1) into LDS (async b128 -> LDS) ------------------
    asyncStageH(Hbf + ((t + 1) & 1) * (BATCH * HID), hb, tid);
    __syncthreads();
  }

  // ---- persist c slice for next layer (carry-over, no reset) ---------------
#pragma unroll
  for (int r = 0; r < 4; ++r) {
    int idx = tid + r * 256;
    int m = idx >> 5, j = idx & 31;
    Cst[m * HID + wg * WGCOLS + j] = cb[m * CB_S + j];
  }
}

// ---------------------------------------------------------------------------
// host launcher
// ---------------------------------------------------------------------------
extern "C" void kernel_launch(void* const* d_in, const int* in_sizes, int n_in,
                              void* d_out, int out_size, void* d_ws, size_t ws_size,
                              hipStream_t stream) {
  const float* x   = (const float*)d_in[0];   // [B,S,D]
  const float* Wxw = (const float*)d_in[1];   // [L,4H,D]
  const float* Wxb = (const float*)d_in[2];   // [L,4H]
  const float* Whw = (const float*)d_in[3];   // [L,4H,H]
  const float* Whb = (const float*)d_in[4];   // [L,4H]
  const float* bs  = (const float*)d_in[5];   // [L,4H]
  float* out = (float*)d_out;                 // [B,S,H]

  char* ws = (char*)d_ws;
  size_t offXG  = 0;
  size_t offA   = offXG + (size_t)SEQ * BATCH * GATES * 4;   // xg  (transposed)
  size_t offB   = offA  + (size_t)BATCH * SEQ * HID * 4;     // layer-out ping
  size_t offC   = offB  + (size_t)BATCH * SEQ * HID * 4;     // layer-out pong
  size_t offH   = offC  + (size_t)BATCH * HID * 4;           // c state
  size_t offBar = offH  + (size_t)2 * BATCH * HID * 2;       // h bf16 ping-pong

  float*          xg     = (float*)(ws + offXG);
  float*          bufA   = (float*)(ws + offA);
  float*          bufB   = (float*)(ws + offB);
  float*          cstate = (float*)(ws + offC);
  unsigned short* hbf    = (unsigned short*)(ws + offH);
  unsigned*       bar    = (unsigned*)(ws + offBar);

  const int n2 = 2 * BATCH * HID;
  lstm_init_state<<<(n2 + 255) / 256, 256, 0, stream>>>(cstate, hbf, bar);

  (void)hipFuncSetAttribute(reinterpret_cast<const void*>(lstm_scan),
                            hipFuncAttributeMaxDynamicSharedMemorySize, SCAN_LDS);

  const float* layerIn[LAYERS]  = { x, bufA, bufB };
  float*       layerOut[LAYERS] = { bufA, bufB, out };

  for (int l = 0; l < LAYERS; ++l) {
    dim3 g(GATES / 128, (BATCH * SEQ) / 64);
    lstm_xgemm<<<g, 256, 0, stream>>>(layerIn[l],
                                      Wxw + (size_t)l * GATES * DIN,
                                      Wxb + (size_t)l * GATES,
                                      bs  + (size_t)l * GATES,
                                      xg);
    lstm_scan<<<NWG, 256, SCAN_LDS, stream>>>(xg,
                                              Whw + (size_t)l * GATES * HID,
                                              Whb + (size_t)l * GATES,
                                              layerOut[l], cstate, hbf, bar);
  }
}
